// CompressionSDF_54580444398174
// MI455X (gfx1250) — matrix operand order repaired and checked by reading, not compiled, not run on hardware
//
#include <hip/hip_runtime.h>
#include <hip/hip_bf16.h>

typedef __attribute__((ext_vector_type(16))) _Float16 v16h;
typedef __attribute__((ext_vector_type(8)))  _Float16 v8h;
typedef __attribute__((ext_vector_type(2)))  _Float16 h2;
typedef __attribute__((ext_vector_type(8)))  float    v8f;

// LeakyReLU(0.01): slope < 1 so lrelu(x) == max(x, 0.01*x)  (2 VALU ops)
#define LRELU(v) fmaxf((v), 0.01f * (v))

__device__ __forceinline__ v8f wmma16x16x32(v16h a, v16h b, v8f c) {
  // D(f32 16x16) = A(f16 16x32) * B(f16 32x16) + C   (C carries the bias)
  return __builtin_amdgcn_wmma_f32_16x16x32_f16(false, a, false, b, (short)0, c,
                                                false, false);
}

// C tiles (f32, M=channel, N=voxel) -> f16 B operand (K=32 x N=16) for next
// WMMA. clo holds channels [0..15] of a 32-channel block, chi holds [16..31].
// Pack f32 pairs to f16x2 first, then shuffle packed words: 8 shfl + 8 selects.
__device__ __forceinline__ v16h relayout32(v8f clo, v8f chi, int hi) {
  union V16 { float f[8]; v16h h; } b;
#pragma unroll
  for (int j = 0; j < 4; ++j) {
    h2 plo; plo[0] = (_Float16)clo[2 * j]; plo[1] = (_Float16)clo[2 * j + 1];
    h2 phi; phi[0] = (_Float16)chi[2 * j]; phi[1] = (_Float16)chi[2 * j + 1];
    float plo_f = __builtin_bit_cast(float, plo);
    float phi_f = __builtin_bit_cast(float, phi);
    float plo_o = __shfl_xor(plo_f, 16, 32);   // partner half-wave's packed pair
    float phi_o = __shfl_xor(phi_f, 16, 32);
    b.f[j]     = hi ? phi_o : plo_f;  // elements 2j..2j+1   (K = hi*16 + 2j..)
    b.f[j + 4] = hi ? phi_f : plo_o;  // elements 2j+8..2j+9 (K = hi*16+8+2j..)
  }
  return b.h;
}

// ---------------------------------------------------------------------------
// Kernel A: 1x1 conv stack 128->64->32->16 over 65536 pixels (4096 groups of
// 16). Weights staged in LDS in A-fragment layout; f written to ws as f16 in
// kernel-B's B-operand-friendly layout: ws[g*256 + pixel*16 + channel].
// ---------------------------------------------------------------------------
__global__ __launch_bounds__(256)
void sdf_stage_kernel(const float* __restrict__ x,
                      const float* __restrict__ sw1, const float* __restrict__ sb1,
                      const float* __restrict__ sw2, const float* __restrict__ sb2,
                      const float* __restrict__ sw3, const float* __restrict__ sb3,
                      _Float16* __restrict__ fws) {
  __shared__ v16h ldsA[21 * 32];  // 16 tiles s1, 4 tiles s2, 1 tile s3
  const int tid  = threadIdx.x;
  const int lane = tid & 31, n = lane & 15, hi = lane >> 4;

  // Cooperative one-time weight staging into WMMA A-fragment layout.
  for (int idx = tid; idx < 21 * 32; idx += 256) {
    int t = idx >> 5, l = idx & 31;
    int ln = l & 15, lhi = l >> 4;
    const float* W; int Cin, m0, kk;
    if (t < 16)      { W = sw1; Cin = 128; m0 = (t >> 2) << 4; kk = (t & 3) << 5; }
    else if (t < 20) { int u = t - 16; W = sw2; Cin = 64; m0 = (u >> 1) << 4; kk = (u & 1) << 5; }
    else             { W = sw3; Cin = 32; m0 = 0; kk = 0; }
    int m = m0 + ln;
    v16h a;
#pragma unroll
    for (int e = 0; e < 16; ++e) {
      int K = kk + lhi * 16 + e;
      a[e] = (_Float16)W[m * Cin + K];
    }
    ldsA[idx] = a;
  }
  __syncthreads();

  const int g   = blockIdx.x * 8 + (tid >> 5);  // 512 blocks * 8 waves = 4096
  const int b   = g >> 10;
  const int pix = (g & 1023) << 4;
  const float* xb = x + (size_t)b * 128 * 16384 + pix;

  // B operand: x channels, K=128 in 4 chunks of 32
  v16h Bx[4];
#pragma unroll
  for (int kc = 0; kc < 4; ++kc)
#pragma unroll
    for (int e = 0; e < 16; ++e) {
      int c = kc * 32 + hi * 16 + e;
      Bx[kc][e] = (_Float16)xb[(size_t)c * 16384 + n];
    }

  // layer s1: 64x128, bias carried in C operand
  v8f acc[4];
#pragma unroll
  for (int mt = 0; mt < 4; ++mt) {
#pragma unroll
    for (int r = 0; r < 8; ++r) acc[mt][r] = sb1[mt * 16 + hi * 8 + r];
#pragma unroll
    for (int kc = 0; kc < 4; ++kc)
      acc[mt] = wmma16x16x32(ldsA[(mt * 4 + kc) * 32 + lane], Bx[kc], acc[mt]);
#pragma unroll
    for (int r = 0; r < 8; ++r) acc[mt][r] = LRELU(acc[mt][r]);
  }
  v16h b2[2];
  b2[0] = relayout32(acc[0], acc[1], hi);
  b2[1] = relayout32(acc[2], acc[3], hi);

  // layer s2: 32x64
  v8f acc2[2];
#pragma unroll
  for (int mt = 0; mt < 2; ++mt) {
#pragma unroll
    for (int r = 0; r < 8; ++r) acc2[mt][r] = sb2[mt * 16 + hi * 8 + r];
#pragma unroll
    for (int kc = 0; kc < 2; ++kc)
      acc2[mt] = wmma16x16x32(ldsA[(16 + mt * 2 + kc) * 32 + lane], b2[kc], acc2[mt]);
#pragma unroll
    for (int r = 0; r < 8; ++r) acc2[mt][r] = LRELU(acc2[mt][r]);
  }
  v16h b3 = relayout32(acc2[0], acc2[1], hi);

  // layer s3: 16x32, no activation, bias in C operand
  v8f c3init;
#pragma unroll
  for (int r = 0; r < 8; ++r) c3init[r] = sb3[hi * 8 + r];
  v8f c3 = wmma16x16x32(ldsA[20 * 32 + lane], b3, c3init);

  v8h st;
#pragma unroll
  for (int r = 0; r < 8; ++r) st[r] = (_Float16)c3[r];
  // lane n (hi=0): channels 0..7 of pixel n; lane n+16: channels 8..15
  *(v8h*)(fws + (size_t)g * 256 + n * 16 + hi * 8) = st;
}

// ---------------------------------------------------------------------------
// Kernel B: per-voxel MLP 17->32->32->16->1 + sigmoid. One wave = one 16-pixel
// group x 32 depth slices; feature half of layer-1 B operand is z-invariant.
// ---------------------------------------------------------------------------
__global__ __launch_bounds__(256)
void sdf_mlp_kernel(const _Float16* __restrict__ fws,
                    const float* __restrict__ mw1, const float* __restrict__ mb1,
                    const float* __restrict__ mw2, const float* __restrict__ mb2,
                    const float* __restrict__ mw3, const float* __restrict__ mb3,
                    const float* __restrict__ mw4, const float* __restrict__ mb4,
                    float* __restrict__ out) {
  const int tid  = threadIdx.x;
  const int lane = tid & 31, n = lane & 15, hi = lane >> 4;
  const int task = blockIdx.x * 8 + (tid >> 5);  // 2048 blocks * 8 = 16384
  const int g    = task >> 2;
  const int zc   = task & 3;
  const int b    = g >> 10;
  const int pix  = (g & 1023) << 4;

  // MLP weights -> registers in A-fragment layout (layer-1 K padded 17->32).
  v16h a1[2], a2[2], a3;
#pragma unroll
  for (int mt = 0; mt < 2; ++mt) {
    int m = mt * 16 + n;
#pragma unroll
    for (int e = 0; e < 16; ++e) {
      int K = hi * 16 + e;
      a1[mt][e] = (K < 17) ? (_Float16)mw1[m * 17 + K] : (_Float16)0.f;
      a2[mt][e] = (_Float16)mw2[m * 32 + K];
      if (mt == 0) a3[e] = (_Float16)mw3[n * 32 + K];
    }
  }
  // biases in C-operand layout (per-lane v8f), layer-4 weights per lane half
  v8f bs1[2], bs2[2], bs3, w4l;
#pragma unroll
  for (int r = 0; r < 8; ++r) {
    bs1[0][r] = mb1[hi * 8 + r];      bs1[1][r] = mb1[16 + hi * 8 + r];
    bs2[0][r] = mb2[hi * 8 + r];      bs2[1][r] = mb2[16 + hi * 8 + r];
    bs3[r]    = mb3[hi * 8 + r];
    w4l[r]    = mw4[hi * 8 + r];
  }
  const float b4 = mb4[0];

  // z-invariant half of the layer-1 B operand: 16 feature channels, one 32B load.
  v16h bload = *(const v16h*)(fws + (size_t)g * 256 + n * 16);
  v16h b1;
#pragma unroll
  for (int e = 0; e < 16; ++e) b1[e] = hi ? (_Float16)0.f : bload[e];

  float* op = out + (size_t)b * 2097152 + pix + n;

  for (int zi = 0; zi < 32; ++zi) {
    const int z = zc * 32 + zi;
    const float zf = (2.f * (float)z) / 127.f - 1.f;  // linspace(-1,1,128)
    v16h b1c = b1;
    b1c[0] = hi ? (_Float16)zf : b1c[0];  // K=16: coordinate channel

    // layer 1: 32x17(pad 32), bias carried in the C operand
    v8f c1a = wmma16x16x32(a1[0], b1c, bs1[0]);
    v8f c1b = wmma16x16x32(a1[1], b1c, bs1[1]);
#pragma unroll
    for (int r = 0; r < 8; ++r) { c1a[r] = LRELU(c1a[r]); c1b[r] = LRELU(c1b[r]); }
    v16h b2 = relayout32(c1a, c1b, hi);

    // layer 2: 32x32
    v8f c2a = wmma16x16x32(a2[0], b2, bs2[0]);
    v8f c2b = wmma16x16x32(a2[1], b2, bs2[1]);
#pragma unroll
    for (int r = 0; r < 8; ++r) { c2a[r] = LRELU(c2a[r]); c2b[r] = LRELU(c2b[r]); }
    v16h b3 = relayout32(c2a, c2b, hi);

    // layer 3: 16x32, bias in C operand
    v8f c3 = wmma16x16x32(a3, b3, bs3);

    // layer 4: 1x16 dot + sigmoid (split across wave halves)
    float part = 0.f;
#pragma unroll
    for (int r = 0; r < 8; ++r) part += LRELU(c3[r]) * w4l[r];
    float tot = part + __shfl_xor(part, 16, 32);
    float res = 1.f / (1.f + __expf(-(tot + b4)));
    if (hi == 0) op[(size_t)z * 16384] = res;
  }
}

extern "C" void kernel_launch(void* const* d_in, const int* in_sizes, int n_in,
                              void* d_out, int out_size, void* d_ws, size_t ws_size,
                              hipStream_t stream) {
  (void)in_sizes; (void)n_in; (void)out_size; (void)ws_size;
  const float* x   = (const float*)d_in[0];
  const float* sw1 = (const float*)d_in[1];
  const float* sb1 = (const float*)d_in[2];
  const float* sw2 = (const float*)d_in[3];
  const float* sb2 = (const float*)d_in[4];
  const float* sw3 = (const float*)d_in[5];
  const float* sb3 = (const float*)d_in[6];
  const float* mw1 = (const float*)d_in[7];
  const float* mb1 = (const float*)d_in[8];
  const float* mw2 = (const float*)d_in[9];
  const float* mb2 = (const float*)d_in[10];
  const float* mw3 = (const float*)d_in[11];
  const float* mb3 = (const float*)d_in[12];
  const float* mw4 = (const float*)d_in[13];
  const float* mb4 = (const float*)d_in[14];
  _Float16* fws = (_Float16*)d_ws;   // 4096 groups * 256 halves = 2 MB
  float* out = (float*)d_out;

  hipLaunchKernelGGL(sdf_stage_kernel, dim3(512), dim3(256), 0, stream,
                     x, sw1, sb1, sw2, sb2, sw3, sb3, fws);
  hipLaunchKernelGGL(sdf_mlp_kernel, dim3(2048), dim3(256), 0, stream,
                     fws, mw1, mb1, mw2, mb2, mw3, mb3, mw4, mb4, out);
}
